// Classifier_59270548685016
// MI455X (gfx1250) — matrix operand order repaired and checked by reading, not compiled, not run on hardware
//
#include <hip/hip_runtime.h>

// prediction[256,1000] = x[256,D] . c[1000,D]^T,  D = 3*224*224 = 150528
//
// Pipeline (all deterministic, no atomics):
//   1) split_x_kernel : x fp32 -> bf16 hi/lo in WMMA-fragment-swizzled layout (d_ws)
//   2) gemm_kernel    : split-bf16 WMMA (hi*hi + lo*hi + hi*lo), c converted on the fly
//                       with a hand-forced 2-ops/element split, split-K partials to d_ws
//   3) reduce_kernel  : sum 48 split-K partials -> d_out

#define D_DIM   150528L
#define M_DIM   256
#define N_DIM   1000
#define NTILE   64                    // output columns per workgroup (4 waves x 64x64 tile)
#define KSTEP   32                    // K per WMMA
#define KBLKS   (150528 / 32)         // 4704 k-blocks
#define SPLITK  48                    // split-K slices (48 * 98 = 4704)
#define KB_PER_SLICE (KBLKS / SPLITK) // 98
#define NPAD    1024                  // padded N for partials

typedef __attribute__((ext_vector_type(16))) __bf16   v16bf;
typedef __attribute__((ext_vector_type(8)))  float    v8f;
typedef __attribute__((ext_vector_type(4)))  float    f32x4;
typedef __attribute__((ext_vector_type(16))) float    f32x16;
typedef __attribute__((ext_vector_type(8)))  unsigned u32x8;

// Optimal split of 2 fp32 -> packed bf16 hi pair + packed bf16 lo pair (4 VALU total):
//   h  = v_cvt_pk_bf16_f32(f0, f1)
//   l0 = v_fma_mix_f32_bf16(h.lo16, -1.0, f0)   // f0 - (float)hi0, exact in fp32
//   l1 = v_fma_mix_f32_bf16(h.hi16, -1.0, f1)
//   l  = v_cvt_pk_bf16_f32(l0, l1)
__device__ __forceinline__ void split2(float f0, float f1, unsigned& hp, unsigned& lp) {
    unsigned h;
    float l0, l1;
    asm("v_cvt_pk_bf16_f32 %0, %1, %2" : "=v"(h) : "v"(f0), "v"(f1));
    asm("v_fma_mix_f32_bf16 %0, %1, -1.0, %2 op_sel:[0,1,0] op_sel_hi:[1,1,0]"
        : "=v"(l0) : "v"(h), "v"(f0));
    asm("v_fma_mix_f32_bf16 %0, %1, -1.0, %2 op_sel:[1,1,0] op_sel_hi:[1,1,0]"
        : "=v"(l1) : "v"(h), "v"(f1));
    asm("v_cvt_pk_bf16_f32 %0, %1, %2" : "=v"(lp) : "v"(l0), "v"(l1));
    hp = h;
}

__device__ __forceinline__ void split_frag16(const f32x16& fv, v16bf& hi, v16bf& lo) {
    u32x8 h, l;
#pragma unroll
    for (int j = 0; j < 8; ++j) {
        unsigned hp, lp;
        split2(fv[2 * j], fv[2 * j + 1], hp, lp);
        h[j] = hp; l[j] = lp;
    }
    hi = __builtin_bit_cast(v16bf, h);
    lo = __builtin_bit_cast(v16bf, l);
}

// ---------------------------------------------------------------------------
// 1) x fp32 -> fragment-swizzled bf16 hi/lo.
// Fragment block = 16 rows x 32 k, lane-major (32B per lane) in the exact
// A-operand order of V_WMMA_F32_16X16X32_BF16:
//   lanes 0-15 : row = L,    K = {0..7, 16..23}
//   lanes 16-31: row = L-16, K = {8..15, 24..31}
// ---------------------------------------------------------------------------
__global__ void __launch_bounds__(256)
split_x_kernel(const float* __restrict__ X,
               __bf16* __restrict__ XH, __bf16* __restrict__ XL)
{
    const int lane  = threadIdx.x & 31;
    const int wave  = threadIdx.x >> 5;           // 0..7
    const int k_blk = blockIdx.x * 8 + wave;      // 0..4703
    const int m_blk = blockIdx.y;                 // 0..15
    const int r     = lane & 15;
    const int hi16  = lane >> 4;

    const long kb  = (long)k_blk * 32 + hi16 * 8;
    const long row = (long)m_blk * 16 + r;
    const float* p = X + row * D_DIM + kb;

    f32x4 s0 = *(const f32x4*)(p + 0);
    f32x4 s1 = *(const f32x4*)(p + 4);
    f32x4 s2 = *(const f32x4*)(p + 16);
    f32x4 s3 = *(const f32x4*)(p + 20);
    f32x16 fv;
#pragma unroll
    for (int i = 0; i < 4; ++i) {
        fv[i] = s0[i]; fv[4 + i] = s1[i]; fv[8 + i] = s2[i]; fv[12 + i] = s3[i];
    }

    v16bf hi, lo;
    split_frag16(fv, hi, lo);

    const long blk_off = ((long)m_blk * KBLKS + k_blk) * 512 + lane * 16;  // bf16 units
    *(v16bf*)(XH + blk_off) = hi;
    *(v16bf*)(XL + blk_off) = lo;
}

// ---------------------------------------------------------------------------
// 2) GEMM: 4 waves per WG covering full M=256 (wave w -> rows 64w..64w+63),
// wave tile 64x64 = 4x4 grid of 16x16 accumulators; N-tile 64; split-K over 48.
// A comes pre-split from d_ws (pure b128 loads, zero conversion VALU);
// B (c) is streamed non-temporally and split on the fly with split2().
// ---------------------------------------------------------------------------
__global__ void __launch_bounds__(128)
gemm_kernel(const __bf16* __restrict__ XH, const __bf16* __restrict__ XL,
            const float* __restrict__ C, float* __restrict__ P)
{
    const int lane = threadIdx.x & 31;
    const int wave = threadIdx.x >> 5;
    const int hi16 = lane >> 4;
    const int r    = lane & 15;

    const int n0     = blockIdx.x * NTILE;
    const int slice  = blockIdx.y;
    const int kb_beg = slice * KB_PER_SLICE;
    const int kb_end = kb_beg + KB_PER_SLICE;

    v8f acc[4][4];
#pragma unroll
    for (int mb = 0; mb < 4; ++mb)
#pragma unroll
        for (int nb = 0; nb < 4; ++nb)
#pragma unroll
            for (int v = 0; v < 8; ++v) acc[mb][nb][v] = 0.0f;

    // A fragment streams: contiguous 1KB block per k-step.
    const __bf16* ah_ptr[4];
    const __bf16* al_ptr[4];
#pragma unroll
    for (int mb = 0; mb < 4; ++mb) {
        const long base = ((long)(wave * 4 + mb) * KBLKS + kb_beg) * 512 + lane * 16;
        ah_ptr[mb] = XH + base;
        al_ptr[mb] = XL + base;
    }

    // B operand (32x16): lane r = column of B = row of c;
    // lanes 0-15 hold K={0..7,16..23}, lanes 16-31 hold K={8..15,24..31} (matches A).
    const float* crow[4];
#pragma unroll
    for (int nb = 0; nb < 4; ++nb) {
        int n = n0 + nb * 16 + r;
        if (n >= N_DIM) n = N_DIM - 1;            // clamp (cols >=1000 never reduced)
        crow[nb] = C + (long)n * D_DIM + hi16 * 8;
    }

    for (int kb = kb_beg; kb < kb_end; ++kb) {
        const long koff = (long)kb * 32;

        // All 4 waves read the same c rows: one wave prefetches for the WG.
        if (wave == 0) {
#pragma unroll
            for (int nb = 0; nb < 4; ++nb)
                __builtin_prefetch(crow[nb] + koff + 2048, 0, 1);   // 8KB ahead
        }

        // A: bf16 already split -> pure b128 loads.
        v16bf ah[4], al[4];
#pragma unroll
        for (int mb = 0; mb < 4; ++mb) {
            ah[mb] = *(const v16bf*)ah_ptr[mb];
            al[mb] = *(const v16bf*)al_ptr[mb];
            ah_ptr[mb] += 512;
            al_ptr[mb] += 512;
        }

#pragma unroll
        for (int nb = 0; nb < 4; ++nb) {
            const float* p = crow[nb] + koff;
            f32x4 s0 = __builtin_nontemporal_load((const f32x4*)(p + 0));
            f32x4 s1 = __builtin_nontemporal_load((const f32x4*)(p + 4));
            f32x4 s2 = __builtin_nontemporal_load((const f32x4*)(p + 16));
            f32x4 s3 = __builtin_nontemporal_load((const f32x4*)(p + 20));
            f32x16 fv;
#pragma unroll
            for (int i = 0; i < 4; ++i) {
                fv[i] = s0[i]; fv[4 + i] = s1[i]; fv[8 + i] = s2[i]; fv[12 + i] = s3[i];
            }
            v16bf bh, bl;
            split_frag16(fv, bh, bl);

            // D = A*B + C with split-bf16: hi*hi + lo*hi + hi*lo.
#pragma unroll
            for (int mb = 0; mb < 4; ++mb) {
                acc[mb][nb] = __builtin_amdgcn_wmma_f32_16x16x32_bf16(
                    false, ah[mb], false, bh, (short)0, acc[mb][nb], false, false);
                acc[mb][nb] = __builtin_amdgcn_wmma_f32_16x16x32_bf16(
                    false, al[mb], false, bh, (short)0, acc[mb][nb], false, false);
                acc[mb][nb] = __builtin_amdgcn_wmma_f32_16x16x32_bf16(
                    false, ah[mb], false, bl, (short)0, acc[mb][nb], false, false);
            }
        }
    }

    // Partials: P[slice][m][n] (n padded to 1024). C/D layout: N = lane%16,
    // VGPR v -> M = v + 8*(lane>=16). Plain stores, no atomics.
    const int m_off = hi16 * 8;
#pragma unroll
    for (int mb = 0; mb < 4; ++mb)
#pragma unroll
        for (int nb = 0; nb < 4; ++nb) {
            const int n = n0 + nb * 16 + r;
#pragma unroll
            for (int v = 0; v < 8; ++v) {
                const int m = wave * 64 + mb * 16 + m_off + v;
                P[((long)slice * M_DIM + m) * NPAD + n] = acc[mb][nb][v];
            }
        }
}

// ---------------------------------------------------------------------------
// 3) Deterministic split-K reduction -> d_out [256,1000].
// ---------------------------------------------------------------------------
__global__ void __launch_bounds__(256)
reduce_kernel(const float* __restrict__ P, float* __restrict__ Out, int total)
{
    const int i = blockIdx.x * blockDim.x + threadIdx.x;
    if (i >= total) return;
    const int m = i / N_DIM;
    const int n = i - m * N_DIM;
    float s = 0.0f;
#pragma unroll
    for (int j = 0; j < SPLITK; ++j)
        s += P[((long)j * M_DIM + m) * NPAD + n];
    Out[i] = s;
}

extern "C" void kernel_launch(void* const* d_in, const int* in_sizes, int n_in,
                              void* d_out, int out_size, void* d_ws, size_t ws_size,
                              hipStream_t stream) {
    const float* X = (const float*)d_in[0];   // [256, 3, 224, 224]
    const float* C = (const float*)d_in[1];   // [1000, 3, 224, 224]
    float* Out = (float*)d_out;               // [256, 1000] fp32

    const long x_elems = (long)M_DIM * D_DIM;           // 38,535,168
    __bf16* XH = (__bf16*)d_ws;
    __bf16* XL = XH + x_elems;
    float*  P  = (float*)((char*)d_ws + 2 * x_elems * sizeof(__bf16));  // +154MB

    // 1) pre-split x into fragment-swizzled bf16 hi/lo  (~308MB traffic, ~13us)
    split_x_kernel<<<dim3(KBLKS / 8, 16), 256, 0, stream>>>(X, XH, XL);

    // 2) WMMA GEMM: 16 n-tiles x 48 split-K slices = 768 workgroups (3072 waves)
    gemm_kernel<<<dim3((N_DIM + NTILE - 1) / NTILE, SPLITK), 128, 0, stream>>>(XH, XL, C, P);

    // 3) deterministic reduction of the 48 partial slices
    reduce_kernel<<<(out_size + 255) / 256, 256, 0, stream>>>(P, Out, out_size);
}
